// Attention_81054622810752
// MI455X (gfx1250) — matrix operand order repaired and checked by reading, not compiled, not run on hardware
//
#include <hip/hip_runtime.h>
#include <hip/hip_bf16.h>

// MI455X / gfx1250 fused multi-head attention (QKV proj + flash attention).
// Needs ws_size >= 24 MB (Q, K, V^T in f16).

typedef __attribute__((ext_vector_type(16))) _Float16 v16h;
typedef __attribute__((ext_vector_type(8)))  _Float16 v8h;
typedef __attribute__((ext_vector_type(8)))  float    v8f;

#define WMMA_F32_F16(a, b, c) \
  __builtin_amdgcn_wmma_f32_16x16x32_f16(false, (a), false, (b), (short)0, (c), false, false)

static __device__ __forceinline__ v8h ld8h(const _Float16* p) {
  v8h r;
  __builtin_memcpy(&r, __builtin_assume_aligned(p, 16), 16);
  return r;
}
static __device__ __forceinline__ v16h mk16(v8h lo, v8h hi) {
  v16h a;
#pragma unroll
  for (int i = 0; i < 8; ++i) { a[i] = lo[i]; a[i + 8] = hi[i]; }
  return a;
}
static __device__ __forceinline__ void st4h(_Float16* p, float4 v) {
  _Float16 t[4] = { (_Float16)v.x, (_Float16)v.y, (_Float16)v.z, (_Float16)v.w };
  __builtin_memcpy(__builtin_assume_aligned(p, 8), t, 8);
}

// -------------------------------------------------------------------------
// Kernel 1: QKV projection. x[4096,1024] @ W[1024,3072] + b.
// Output: Q (x SCALE), K as f16 [bh][n][64]; V transposed f16 [bh][64][n].
// Workgroup tile 128x64, 8 waves, each wave a 32x32 patch, K-step 32.
// LDS row stride padded to 40 halves (20 dwords) -> conflict-free frag reads.
// Epilogue routing (q/k/v, head, batch) is computed from fragment-base
// scalars so the branches are wave-uniform (s_cbranch, no exec divergence).
// -------------------------------------------------------------------------
__global__ __launch_bounds__(256) void qkv_proj_kernel(
    const float* __restrict__ x, const float* __restrict__ W,
    const float* __restrict__ bqkv,
    _Float16* __restrict__ qh, _Float16* __restrict__ kh,
    _Float16* __restrict__ vth)
{
  __shared__ __align__(16) _Float16 Al[128 * 40];
  __shared__ __align__(16) _Float16 Bl[64 * 40];

  const int t    = threadIdx.x;
  const int lane = t & 31, wid = t >> 5;
  const int l16  = lane & 15, lh = lane >> 4;
  const int wr   = (wid & 3) * 32;   // wave row offset in tile
  const int wc   = (wid >> 2) * 32;  // wave col offset in tile
  const int tm0  = blockIdx.x * 128;
  const int tn0  = blockIdx.y * 64;

  v8f acc[2][2];
#pragma unroll
  for (int i = 0; i < 2; ++i)
#pragma unroll
    for (int j = 0; j < 2; ++j) acc[i][j] = {};

  const float4* x4 = (const float4*)x;
  const float4* w4 = (const float4*)W;

  for (int k0 = 0; k0 < 1024; k0 += 32) {
    // stage A (x) tile 128x32 fp32 -> f16 LDS
#pragma unroll
    for (int it = 0; it < 4; ++it) {
      int f = t + it * 256;
      int row = f >> 3, c4 = f & 7;
      float4 v = x4[(size_t)(tm0 + row) * 256 + (k0 >> 2) + c4];
      st4h(&Al[row * 40 + c4 * 4], v);
    }
    // stage B (W) tile 32x64 fp32 -> f16 LDS, transposed to [n][k]
#pragma unroll
    for (int it = 0; it < 2; ++it) {
      int f = t + it * 256;
      int kr = f >> 4, c4 = f & 15;
      float4 v = w4[(size_t)(k0 + kr) * 768 + (tn0 >> 2) + c4];
      int n0 = c4 * 4;
      Bl[(n0 + 0) * 40 + kr] = (_Float16)v.x;
      Bl[(n0 + 1) * 40 + kr] = (_Float16)v.y;
      Bl[(n0 + 2) * 40 + kr] = (_Float16)v.z;
      Bl[(n0 + 3) * 40 + kr] = (_Float16)v.w;
    }
    __syncthreads();

    // A frag 16x32: lanes 0-15 hold K {0..7,16..23}, lanes 16-31 {8..15,24..31}
    v16h afrag[2], bfrag[2];
#pragma unroll
    for (int rs = 0; rs < 2; ++rs) {
      const _Float16* pa = &Al[(wr + rs * 16 + l16) * 40 + 8 * lh];
      afrag[rs] = mk16(ld8h(pa), ld8h(pa + 16));
    }
    // B frag 32x16: lanes 0-15 hold K {0..7,16..23}, lanes 16-31 {8..15,24..31}
#pragma unroll
    for (int cs = 0; cs < 2; ++cs) {
      const _Float16* pb = &Bl[(wc + cs * 16 + l16) * 40 + 8 * lh];
      bfrag[cs] = mk16(ld8h(pb), ld8h(pb + 16));
    }
#pragma unroll
    for (int rs = 0; rs < 2; ++rs)
#pragma unroll
      for (int cs = 0; cs < 2; ++cs)
        acc[rs][cs] = WMMA_F32_F16(afrag[rs], bfrag[cs], acc[rs][cs]);
    __syncthreads();
  }

  // epilogue: bias + route to Q (scaled), K, V^T. qkv / head / batch are
  // uniform per fragment (16-col / 16-row tiles never straddle 64/1024/2048
  // boundaries), so branches below are wave-uniform.
#pragma unroll
  for (int cs = 0; cs < 2; ++cs) {
    const int cbase = tn0 + wc + cs * 16;     // multiple of 16, scalar
    const int qkv   = cbase >> 10;            // 0/1/2, scalar
    const int hh    = (cbase >> 6) & 15;      // head, scalar
    const int d     = (cbase & 63) + l16;     // 0..63
    const float bias = bqkv[cbase + l16];

    if (qkv == 2) {
#pragma unroll
      for (int rs = 0; rs < 2; ++rs) {
        const int rbase = tm0 + wr + rs * 16;          // scalar
        const int bi = rbase >> 11, nbase = rbase & 2047;
        const size_t base = ((size_t)(bi * 16 + hh) * 64 + d) * 2048 + nbase;
#pragma unroll
        for (int r = 0; r < 8; ++r)
          vth[base + r + 8 * lh] = (_Float16)(acc[rs][cs][r] + bias);
      }
    } else {
      _Float16* dst = (qkv == 0) ? qh : kh;
      const float sc = (qkv == 0) ? 0.125f : 1.0f;     // D^-0.5 folded into Q
#pragma unroll
      for (int rs = 0; rs < 2; ++rs) {
        const int rbase = tm0 + wr + rs * 16;
        const int bi = rbase >> 11, nbase = rbase & 2047;
        const size_t base = ((size_t)(bi * 16 + hh) * 2048 + nbase) * 64 + d;
#pragma unroll
        for (int r = 0; r < 8; ++r)
          dst[base + (size_t)(r + 8 * lh) * 64] = (_Float16)((acc[rs][cs][r] + bias) * sc);
      }
    }
  }
}

// -------------------------------------------------------------------------
// Kernel 2: flash attention. One wave per (bh, 32-query tile): two 16-row
// tiles share every K/V B-fragment (2x arithmetic intensity on K/V).
// Per 32-key block: 8 WMMAs (S), online softmax with 16-lane shuffles,
// P via per-wave LDS slab into A-layout, 8 WMMAs (P*V) into 32x64 f32 O.
// -------------------------------------------------------------------------
__global__ __launch_bounds__(256) void flash_attn_kernel(
    const _Float16* __restrict__ qh, const _Float16* __restrict__ kh,
    const _Float16* __restrict__ vth, float* __restrict__ out)
{
  __shared__ __align__(16) _Float16 Pl[8 * 32 * 40];   // 20 KB

  const int t    = threadIdx.x;
  const int lane = t & 31, wid = t >> 5;
  const int l16  = lane & 15, lh = lane >> 4;
  const int gw   = blockIdx.x * 8 + wid;  // 0..2047
  const int bh   = gw >> 6;               // 0..31
  const int qbase = (gw & 63) * 32;       // 0..2016
  _Float16* P = &Pl[wid * 1280];

  // Q fragments: 2 row-tiles x 2 K=32 chunks (Q already scaled by D^-0.5)
  v16h aq[2][2];
#pragma unroll
  for (int rt = 0; rt < 2; ++rt) {
    const _Float16* qrow = qh + ((size_t)bh * 2048 + qbase + rt * 16 + l16) * 64;
#pragma unroll
    for (int kc = 0; kc < 2; ++kc) {
      const _Float16* p = qrow + kc * 32 + 8 * lh;
      aq[rt][kc] = mk16(ld8h(p), ld8h(p + 16));
    }
  }

  v8f o[2][4];
#pragma unroll
  for (int rt = 0; rt < 2; ++rt)
#pragma unroll
    for (int f = 0; f < 4; ++f) o[rt][f] = {};
  float mrow[2][8], lrow[2][8];
#pragma unroll
  for (int rt = 0; rt < 2; ++rt)
#pragma unroll
    for (int r = 0; r < 8; ++r) { mrow[rt][r] = -1e30f; lrow[rt][r] = 0.f; }

  const _Float16* Kbh = kh + (size_t)bh * 2048 * 64;
  const _Float16* Vbh = vth + (size_t)bh * 64 * 2048;

  for (int kb = 0; kb < 2048; kb += 32) {
    // S = Q * K^T for 32 keys: each K B-frag feeds both row-tiles
    v8f s[2][2];
#pragma unroll
    for (int rt = 0; rt < 2; ++rt) { s[rt][0] = {}; s[rt][1] = {}; }
#pragma unroll
    for (int kc = 0; kc < 2; ++kc) {
      const _Float16* p0 = Kbh + (size_t)(kb + l16) * 64 + kc * 32 + 8 * lh;
      v16h bk0 = mk16(ld8h(p0), ld8h(p0 + 16));
      const _Float16* p1 = Kbh + (size_t)(kb + 16 + l16) * 64 + kc * 32 + 8 * lh;
      v16h bk1 = mk16(ld8h(p1), ld8h(p1 + 16));
#pragma unroll
      for (int rt = 0; rt < 2; ++rt) {
        s[rt][0] = WMMA_F32_F16(aq[rt][kc], bk0, s[rt][0]);
        s[rt][1] = WMMA_F32_F16(aq[rt][kc], bk1, s[rt][1]);
      }
    }

    // online softmax: rows live per-VGPR, cols across the 16-lane half
#pragma unroll
    for (int rt = 0; rt < 2; ++rt) {
#pragma unroll
      for (int r = 0; r < 8; ++r) {
        float tmax = fmaxf(s[rt][0][r], s[rt][1][r]);
#pragma unroll
        for (int m = 8; m >= 1; m >>= 1) tmax = fmaxf(tmax, __shfl_xor(tmax, m, 16));
        float mn   = fmaxf(mrow[rt][r], tmax);
        float corr = __expf(mrow[rt][r] - mn);
        float p0   = __expf(s[rt][0][r] - mn);
        float p1   = __expf(s[rt][1][r] - mn);
        float rsum = p0 + p1;
#pragma unroll
        for (int m = 8; m >= 1; m >>= 1) rsum += __shfl_xor(rsum, m, 16);
        lrow[rt][r] = lrow[rt][r] * corr + rsum;
        mrow[rt][r] = mn;
#pragma unroll
        for (int f = 0; f < 4; ++f) o[rt][f][r] *= corr;
        int prow = rt * 16 + r + 8 * lh;
        P[prow * 40 + l16]      = (_Float16)p0;
        P[prow * 40 + 16 + l16] = (_Float16)p1;
      }
    }

    // reload P in A-fragment layout (two 16x32 frags, K = key-in-block)
    v16h ap[2];
#pragma unroll
    for (int rt = 0; rt < 2; ++rt) {
      const _Float16* pp = &P[(rt * 16 + l16) * 40 + 8 * lh];
      ap[rt] = mk16(ld8h(pp), ld8h(pp + 16));
    }

    // O += P * V  (V^T layout -> contiguous 16B B-frag loads, shared by rts)
#pragma unroll
    for (int f = 0; f < 4; ++f) {
      const _Float16* p = Vbh + (size_t)(f * 16 + l16) * 2048 + kb + 8 * lh;
      v16h bv = mk16(ld8h(p), ld8h(p + 16));
#pragma unroll
      for (int rt = 0; rt < 2; ++rt)
        o[rt][f] = WMMA_F32_F16(ap[rt], bv, o[rt][f]);
    }
  }

  // out[b, n, h*64 + d]  fp32
  const int b = bh >> 4, h = bh & 15;
#pragma unroll
  for (int rt = 0; rt < 2; ++rt) {
#pragma unroll
    for (int f = 0; f < 4; ++f) {
#pragma unroll
      for (int r = 0; r < 8; ++r) {
        int row = qbase + rt * 16 + r + 8 * lh;
        out[((size_t)b * 2048 + row) * 1024 + h * 64 + f * 16 + l16] =
            o[rt][f][r] / lrow[rt][r];
      }
    }
  }
}

extern "C" void kernel_launch(void* const* d_in, const int* in_sizes, int n_in,
                              void* d_out, int out_size, void* d_ws, size_t ws_size,
                              hipStream_t stream) {
  const float* x = (const float*)d_in[0];   // [2,2048,1024]
  const float* W = (const float*)d_in[1];   // [1024,3072]
  const float* b = (const float*)d_in[2];   // [3072]
  float* out = (float*)d_out;               // [2,2048,1024]

  const size_t PER = (size_t)32 * 2048 * 64;  // 4M f16 elements each
  _Float16* qh  = (_Float16*)d_ws;
  _Float16* kh  = qh + PER;
  _Float16* vth = kh + PER;

  qkv_proj_kernel<<<dim3(32, 48), 256, 0, stream>>>(x, W, b, qh, kh, vth);
  flash_attn_kernel<<<dim3(256), 256, 0, stream>>>(qh, kh, vth, out);
}
